// BahdanauDecoder_22162031247796
// MI455X (gfx1250) — matrix operand order repaired and checked by reading, not compiled, not run on hardware
//
#include <hip/hip_runtime.h>
#include <hip/hip_bf16.h>
#include <math.h>

#define H   1024
#define S   16384
#define O   32000
#define FH  4096      // 4*H
#define IN  33024     // O + H

#define ASTRIDE 1032  // padded bf16 row stride: 2064B -> 4-bank rotate/row, conflict-free b128 reads

typedef __attribute__((ext_vector_type(16))) __bf16          v16bf;
typedef __attribute__((ext_vector_type(16))) unsigned short  v16u;
typedef __attribute__((ext_vector_type(8)))  float           v8f;

__device__ __forceinline__ unsigned short f2bf(float x) {
    // fp32 -> bf16, round-to-nearest-even
    unsigned u = __float_as_uint(x);
    return (unsigned short)((u + 0x7FFFu + ((u >> 16) & 1u)) >> 16);
}
__device__ __forceinline__ float sigmoidf_(float x) {
    return 1.0f / (1.0f + __expf(-x));
}
__device__ __forceinline__ float fast_tanh(float x) {
    // branchless: no EXEC manipulation around the WMMA loop, v_exp + v_rcp only
    float t = __expf(-2.0f * fabsf(x));
    float r = (1.0f - t) * __builtin_amdgcn_rcpf(1.0f + t);
    return copysignf(r, x);
}
__device__ __forceinline__ float wave_sum(float x) {
    #pragma unroll
    for (int m = 16; m >= 1; m >>= 1) x += __shfl_xor(x, m, 32);
    return x;
}

// ---------------------------------------------------------------------------
// 0) zero scratch regions that are accumulated into (scores, context)
// ---------------------------------------------------------------------------
__global__ void k_init(float* __restrict__ ws) {
    int i = blockIdx.x * blockDim.x + threadIdx.x;
    if (i < S + 2 * H) ws[i] = 0.0f;   // scores[16384] + hfc[1024] + context[1024]
}

// ---------------------------------------------------------------------------
// 0b) one-time fp32 -> bf16 copy of W_fc_encoder (4MB -> 2MB, reused 512x)
// ---------------------------------------------------------------------------
__global__ __launch_bounds__(256) void k_wfeconv(const float* __restrict__ Wfe,
                                                 unsigned short* __restrict__ Wfeb) {
    const int i = blockIdx.x * 256 + threadIdx.x;
    Wfeb[i] = f2bf(Wfe[i]);
}

// ---------------------------------------------------------------------------
// 1) hfc[r] = sum_k h0[k] * W_fc_hidden[r][k]       (wave-per-row GEMV, 4 MB)
// ---------------------------------------------------------------------------
__global__ __launch_bounds__(256) void k_hproj(const float* __restrict__ h0,
                                               const float* __restrict__ Wfh,
                                               float* __restrict__ hfc) {
    const int lane = threadIdx.x & 31;
    const int row  = blockIdx.x * 8 + (threadIdx.x >> 5);
    const float* wr = Wfh + (size_t)row * H;
    float acc = 0.f;
    #pragma unroll
    for (int i = 0; i < 8; ++i) {
        const int k = i * 128 + lane * 4;
        float4 a = *(const float4*)&wr[k];
        float4 b = *(const float4*)&h0[k];
        acc += a.x * b.x + a.y * b.y + a.z * b.z + a.w * b.w;
    }
    acc = wave_sum(acc);
    if (lane == 0) hfc[row] = acc;
}

// ---------------------------------------------------------------------------
// 2) fused attention scores:
//    scores[s] += sum_n tanh( (enc @ Wfe^T)[s,n] + hfc[n] ) * w[n]
//    GEMM via v_wmma_f32_16x16x32_bf16; A staged in LDS (bf16, padded stride),
//    B streamed (pre-converted bf16 when PRE, else converted in-register).
//    Block = 256 thr (8 waves) = 32 s-rows; wave -> (s-strip, n-quarter).
// ---------------------------------------------------------------------------
template <bool PRE>
__global__ __launch_bounds__(256) void k_scores(const float* __restrict__ enc,
                                                const float* __restrict__ Wfe,
                                                const unsigned short* __restrict__ Wfeb,
                                                const float* __restrict__ hfc,
                                                const float* __restrict__ aw,
                                                float* __restrict__ scores) {
    __shared__ __align__(16) unsigned short Albs[32 * ASTRIDE];   // ~66 KB bf16 A tile

    const int tid    = threadIdx.x;
    const int s_base = blockIdx.x * 32;

    // stage A: 32 rows x 1024 cols, fp32 -> bf16 (coalesced b32 loads)
    for (int i = tid; i < 32 * H; i += 256) {
        const int r = i >> 10, c = i & (H - 1);
        Albs[r * ASTRIDE + c] = f2bf(enc[(size_t)(s_base + r) * H + c]);
    }
    __syncthreads();

    const int wave   = tid >> 5;
    const int lane   = tid & 31;
    const int strip  = wave & 1;     // 2 strips of 16 s-rows
    const int ngrp   = wave >> 1;    // 4 groups of 16 n-tiles
    const int l15    = lane & 15;
    const int hi     = lane >> 4;    // lane half (ISA 16-bit frag layout)

    const unsigned short* arow = &Albs[(strip * 16 + l15) * ASTRIDE];

    float partial[8];
    #pragma unroll
    for (int v = 0; v < 8; ++v) partial[v] = 0.f;

    union VU  { v16u v; uint4 q[2]; unsigned short u[16]; };
    union CU  { v8f  v; float f[8]; };

    for (int nt = 0; nt < 16; ++nt) {
        const int n0 = (ngrp * 16 + nt) * 16;                // n-tile base
        v8f c = {};

        const float*          browf = Wfe  + (size_t)(n0 + l15) * H;  // lane = N column
        const unsigned short* browb = Wfeb + (size_t)(n0 + l15) * H;

        for (int kt = 0; kt < H; kt += 32) {
            // A fragment (16x32 bf16): lane m holds K = kt + hi*16 + [0,16)
            const int kA = kt + hi * 16;
            VU a;
            a.q[0] = *(const uint4*)&arow[kA];        // ds_load_b128
            a.q[1] = *(const uint4*)&arow[kA + 8];
            // B fragment (32x16 bf16): lane n holds K = kt + hi*16 + [0,16)
            const int kB = kt + hi * 16;
            VU b;
            if (PRE) {
                b.q[0] = *(const uint4*)&browb[kB];       // global_load_b128
                b.q[1] = *(const uint4*)&browb[kB + 8];
            } else {
                float tf[16];
                *(float4*)&tf[0]  = *(const float4*)&browf[kB];
                *(float4*)&tf[4]  = *(const float4*)&browf[kB + 4];
                *(float4*)&tf[8]  = *(const float4*)&browf[kB + 8];
                *(float4*)&tf[12] = *(const float4*)&browf[kB + 12];
                #pragma unroll
                for (int j = 0; j < 16; ++j) b.u[j] = f2bf(tf[j]);
            }

            c = __builtin_amdgcn_wmma_f32_16x16x32_bf16(
                    false, __builtin_bit_cast(v16bf, a.v),
                    false, __builtin_bit_cast(v16bf, b.v),
                    (short)0, c, false, false);
        }

        // epilogue: C element (VGPR v, lane) is (s = strip*16 + v + hi*8, n = n0+l15)
        const float bias = hfc[n0 + l15];
        const float ww   = aw[n0 + l15];
        CU cu; cu.v = c;
        #pragma unroll
        for (int v = 0; v < 8; ++v) partial[v] += fast_tanh(cu.f[v] + bias) * ww;
    }

    // reduce over the 16 N-lanes of each half, then accumulate into scores
    #pragma unroll
    for (int v = 0; v < 8; ++v) {
        float x = partial[v];
        x += __shfl_xor(x, 1, 32);
        x += __shfl_xor(x, 2, 32);
        x += __shfl_xor(x, 4, 32);
        x += __shfl_xor(x, 8, 32);
        if (l15 == 0)
            atomicAdd(&scores[s_base + strip * 16 + hi * 8 + v], x);
    }
}

// ---------------------------------------------------------------------------
// 3) softmax over 16384 scores -> attn (written to d_out slot)
// ---------------------------------------------------------------------------
__global__ __launch_bounds__(1024) void k_softmax(const float* __restrict__ scores,
                                                  float* __restrict__ attn) {
    __shared__ float red[32];
    const int tid = threadIdx.x, lane = tid & 31, wave = tid >> 5;

    float m = -3.4e38f;
    for (int i = tid; i < S; i += 1024) m = fmaxf(m, scores[i]);
    #pragma unroll
    for (int x = 16; x >= 1; x >>= 1) m = fmaxf(m, __shfl_xor(m, x, 32));
    if (lane == 0) red[wave] = m;
    __syncthreads();
    m = red[lane];
    #pragma unroll
    for (int x = 16; x >= 1; x >>= 1) m = fmaxf(m, __shfl_xor(m, x, 32));
    m = __shfl(m, 0, 32);
    __syncthreads();

    float s = 0.f;
    for (int i = tid; i < S; i += 1024) s += __expf(scores[i] - m);
    s = wave_sum(s);
    if (lane == 0) red[wave] = s;
    __syncthreads();
    s = red[lane];
    s = wave_sum(s);
    s = __shfl(s, 0, 32);
    const float inv = 1.0f / s;

    for (int i = tid; i < S; i += 1024) attn[i] = __expf(scores[i] - m) * inv;
}

// ---------------------------------------------------------------------------
// 4) context[h] = sum_s attn[s] * enc[s][h]   (64 MB stream, split over grid)
// ---------------------------------------------------------------------------
__global__ __launch_bounds__(256) void k_context(const float* __restrict__ attn,
                                                 const float* __restrict__ enc,
                                                 float* __restrict__ context) {
    const int h  = blockIdx.y * 256 + threadIdx.x;
    const int s0 = blockIdx.x * 256;
    float acc = 0.f;
    for (int s = s0; s < s0 + 256; ++s)
        acc += attn[s] * enc[(size_t)s * H + h];
    atomicAdd(&context[h], acc);
}

// ---------------------------------------------------------------------------
// 5) LSTM gates: gates[j] = b_ih[j]+b_hh[j] + W_ih[j]·[emb;ctx] + W_hh[j]·h0
//    wave-per-row streaming GEMV over the 541 MB W_ih (the HBM-bound piece)
// ---------------------------------------------------------------------------
__global__ __launch_bounds__(256) void k_gates(const float* __restrict__ emb,
                                               const float* __restrict__ context,
                                               const float* __restrict__ h0,
                                               const float* __restrict__ Wih,
                                               const float* __restrict__ Whh,
                                               const float* __restrict__ bih,
                                               const float* __restrict__ bhh,
                                               float* __restrict__ gates) {
    const int lane = threadIdx.x & 31;
    const int j    = blockIdx.x * 8 + (threadIdx.x >> 5);
    const float* wr = Wih + (size_t)j * IN;
    float acc = 0.f;
    // embedded part: 32000 = 8000 float4
    for (int i = lane; i < 8000; i += 32) {
        float4 a = *(const float4*)&wr[i * 4];
        float4 b = *(const float4*)&emb[i * 4];
        acc += a.x * b.x + a.y * b.y + a.z * b.z + a.w * b.w;
    }
    // context part: 1024 = 256 float4 at row offset 32000 (16B aligned)
    const float* wc = wr + O;
    for (int i = lane; i < 256; i += 32) {
        float4 a = *(const float4*)&wc[i * 4];
        float4 b = *(const float4*)&context[i * 4];
        acc += a.x * b.x + a.y * b.y + a.z * b.z + a.w * b.w;
    }
    // recurrent part
    const float* wh = Whh + (size_t)j * H;
    for (int i = lane; i < 256; i += 32) {
        float4 a = *(const float4*)&wh[i * 4];
        float4 b = *(const float4*)&h0[i * 4];
        acc += a.x * b.x + a.y * b.y + a.z * b.z + a.w * b.w;
    }
    acc = wave_sum(acc);
    if (lane == 0) gates[j] = acc + bih[j] + bhh[j];
}

// ---------------------------------------------------------------------------
// 6) LSTM cell update -> h_new, c_new (d_out slots)
// ---------------------------------------------------------------------------
__global__ __launch_bounds__(256) void k_cell(const float* __restrict__ gates,
                                              const float* __restrict__ c0,
                                              float* __restrict__ h_new,
                                              float* __restrict__ c_new) {
    const int j = blockIdx.x * 256 + threadIdx.x;
    if (j >= H) return;
    const float ig = sigmoidf_(gates[j]);
    const float fg = sigmoidf_(gates[H + j]);
    const float gg = fast_tanh(gates[2 * H + j]);
    const float og = sigmoidf_(gates[3 * H + j]);
    const float cn = fg * c0[j] + ig * gg;
    c_new[j] = cn;
    h_new[j] = og * fast_tanh(cn);
}

// ---------------------------------------------------------------------------
// 7) classifier: out[o] = b_cls[o] + W_cls[o]·h_new   (131 MB stream)
// ---------------------------------------------------------------------------
__global__ __launch_bounds__(256) void k_cls(const float* __restrict__ h_new,
                                             const float* __restrict__ Wcls,
                                             const float* __restrict__ bcls,
                                             float* __restrict__ out) {
    const int lane = threadIdx.x & 31;
    const int row  = blockIdx.x * 8 + (threadIdx.x >> 5);
    const float* wr = Wcls + (size_t)row * H;
    float acc = 0.f;
    #pragma unroll
    for (int i = 0; i < 8; ++i) {
        const int k = i * 128 + lane * 4;
        float4 a = *(const float4*)&wr[k];
        float4 b = *(const float4*)&h_new[k];
        acc += a.x * b.x + a.y * b.y + a.z * b.z + a.w * b.w;
    }
    acc = wave_sum(acc);
    if (lane == 0) out[row] = acc + bcls[row];
}

// ---------------------------------------------------------------------------
extern "C" void kernel_launch(void* const* d_in, const int* in_sizes, int n_in,
                              void* d_out, int out_size, void* d_ws, size_t ws_size,
                              hipStream_t stream) {
    const float* inputs = (const float*)d_in[0];   // [32000]
    const float* h0     = (const float*)d_in[1];   // [1024]
    const float* c0     = (const float*)d_in[2];   // [1024]
    const float* enc    = (const float*)d_in[3];   // [16384,1024]
    const float* Wfh    = (const float*)d_in[4];   // [1024,1024]
    const float* Wfe    = (const float*)d_in[5];   // [1024,1024]
    const float* aw     = (const float*)d_in[6];   // [1,1024]
    const float* Wih    = (const float*)d_in[7];   // [4096,33024]
    const float* Whh    = (const float*)d_in[8];   // [4096,1024]
    const float* bih    = (const float*)d_in[9];   // [4096]
    const float* bhh    = (const float*)d_in[10];  // [4096]
    const float* Wcls   = (const float*)d_in[11];  // [32000,1024]
    const float* bcls   = (const float*)d_in[12];  // [32000]

    float* out    = (float*)d_out;          // [0,32000)   logits
    float* out_h  = out + O;                // [32000,33024) h_new
    float* out_c  = out_h + H;              // [33024,34048) c_new
    float* attn   = out_c + H;              // [34048,50432) attn

    float* ws      = (float*)d_ws;
    float* scores  = ws;                    // 16384
    float* hfc     = ws + S;                // 1024
    float* context = ws + S + H;            // 1024
    float* gates   = ws + S + 2 * H;        // 4096
    unsigned short* Wfeb = (unsigned short*)(ws + S + 2 * H + FH);  // 1M bf16 = 2MB

    const size_t need_bf16 = (size_t)(S + 2 * H + FH) * 4 + (size_t)H * H * 2;
    const bool pre = (ws_size >= need_bf16);

    k_init   <<<dim3((S + 2 * H + 255) / 256), 256, 0, stream>>>(ws);
    if (pre)
        k_wfeconv<<<dim3(H * H / 256), 256, 0, stream>>>(Wfe, Wfeb);
    k_hproj  <<<dim3(H / 8),        256, 0, stream>>>(h0, Wfh, hfc);
    if (pre)
        k_scores<true> <<<dim3(S / 32), 256, 0, stream>>>(enc, Wfe, Wfeb, hfc, aw, scores);
    else
        k_scores<false><<<dim3(S / 32), 256, 0, stream>>>(enc, Wfe, Wfeb, hfc, aw, scores);
    k_softmax<<<dim3(1),           1024, 0, stream>>>(scores, attn);
    k_context<<<dim3(S / 256, H / 256), 256, 0, stream>>>(attn, enc, context);
    k_gates  <<<dim3(FH / 8),       256, 0, stream>>>(inputs, context, h0, Wih, Whh,
                                                      bih, bhh, gates);
    k_cell   <<<dim3(H / 256),      256, 0, stream>>>(gates, c0, out_h, out_c);
    k_cls    <<<dim3(O / 8),        256, 0, stream>>>(out_h, Wcls, bcls, out);
}
